// ConvolutionalLayer_343597383757
// MI455X (gfx1250) — matrix operand order repaired and checked by reading, not compiled, not run on hardware
//
#include <hip/hip_runtime.h>

// y = sum_r (A[r] @ x) @ W[r]^T  ==  sum_r A[r] @ Z[r],  Z[r] = x @ W[r]^T
// R=8, N=4096, IN_F=OUT_F=64.
// Pipeline: (1) zfrag: Z[r]=x@W[r]^T in fp32, stored as bf16 in WMMA B-fragment
// layout; (2) main: stream A (fp32, NT loads), convert to bf16 in registers,
// v_wmma_f32_16x16x32_bf16 with f32 accum, write partials per (r,kc) slice;
// (3) reduce: deterministic sum of 32 partial slices into d_out.
// Workspace: 4 MB (Z frags) + 32 MB (partials) = 37,748,736 bytes.

typedef __bf16 bf16;
typedef __bf16 v16bf __attribute__((ext_vector_type(16)));
typedef float  v8f   __attribute__((ext_vector_type(8)));
typedef float  f4    __attribute__((ext_vector_type(4)));

#define RNUM   8
#define NN     4096
#define INF    64
#define OUTF   64
#define KC     4                 // K-split chunks per relation
#define BM     128               // output rows per block
#define KSTEP  32                // bf16 WMMA K
#define MT     (NN / BM)         // 32 M tiles
#define KB2    (NN / KSTEP)      // 128 k-blocks per relation
#define NTILES (OUTF / 16)       // 4 col tiles
#define NSLICE (RNUM * KC)       // 32 partial slices

// ---------------------------------------------------------------------------
// Kernel 1: Z[r][k][n] = sum_i x[k][i] * W[r][n][i], stored bf16 in the exact
// per-lane B-fragment layout for v_wmma_f32_16x16x32_bf16:
//   fragment id = (r*KB2 + k/32)*NTILES + n/16, 32 lanes * 16 bf16 each.
//   lane = (n%16) + 16*(((k%32) >> 3) & 1); element-in-lane = ((k%32) & 7) + 8*((k%32) >= 16).
// ---------------------------------------------------------------------------
__global__ void zfrag_kernel(const float* __restrict__ x,
                             const float* __restrict__ w,
                             bf16* __restrict__ zf) {
  unsigned t = blockIdx.x * blockDim.x + threadIdx.x;  // over RNUM*NN*OUTF
  unsigned n = t & 63u;
  unsigned k = (t >> 6) & 4095u;
  unsigned r = t >> 18;
  const float* xr = x + (size_t)k * INF;
  const float* wr = w + ((size_t)r * OUTF + n) * INF;
  float acc = 0.0f;
#pragma unroll 8
  for (int i = 0; i < INF; ++i) acc = fmaf(xr[i], wr[i], acc);

  unsigned kb2 = k >> 5, kl = k & 31u;
  unsigned nt  = n >> 4, nl = n & 15u;
  unsigned lane = nl + (((kl >> 3) & 1u) << 4);  // +16 for K in [8,16) and [24,32)
  unsigned elem = (kl & 7u) + ((kl >> 4) << 3);  // K 0-7/8-15 -> 0-7, K 16-23/24-31 -> 8-15
  size_t off = ((size_t)((r * KB2 + kb2) * NTILES + nt)) * 512u + lane * 16u + elem;
  zf[off] = (bf16)acc;
}

// ---------------------------------------------------------------------------
// Kernel 2: main WMMA GEMM. grid = (MT, RNUM, KC), block = 256 (8 waves).
// Wave w: rows m0 = mt*128 + w*16, all 4 col tiles; K loop over NN/KC.
// ---------------------------------------------------------------------------
__global__ void __launch_bounds__(256)
rgcn_wmma_kernel(const float* __restrict__ adj,
                 const bf16* __restrict__ zf,
                 float* __restrict__ partial) {
  const unsigned mt   = blockIdx.x;
  const unsigned r    = blockIdx.y;
  const unsigned kc   = blockIdx.z;
  const unsigned wave = threadIdx.x >> 5;
  const unsigned lane = threadIdx.x & 31u;
  const unsigned m0    = mt * BM + wave * 16u;
  const unsigned khalf = lane >> 4;               // 0: K 0-7/16-23, 1: K 8-15/24-31
  const unsigned arowi = m0 + (lane & 15u);

  const float* arow = adj + ((size_t)r * NN + arowi) * NN;
  const unsigned k_begin = kc * (NN / KC);
  const unsigned k_end   = k_begin + (NN / KC);

  v8f acc0 = {}, acc1 = {}, acc2 = {}, acc3 = {};

#pragma unroll 2
  for (unsigned k0 = k_begin; k0 < k_end; k0 += KSTEP) {
    // ---- A fragment: 16 fp32 per lane (two 8-wide runs), cvt -> bf16
    const float* ap = arow + k0 + khalf * 8u;
    f4 p0 = __builtin_nontemporal_load((const f4*)(ap + 0));
    f4 p1 = __builtin_nontemporal_load((const f4*)(ap + 4));
    f4 p2 = __builtin_nontemporal_load((const f4*)(ap + 16));
    f4 p3 = __builtin_nontemporal_load((const f4*)(ap + 20));
    v16bf a;
    a[0]  = (bf16)p0[0]; a[1]  = (bf16)p0[1]; a[2]  = (bf16)p0[2]; a[3]  = (bf16)p0[3];
    a[4]  = (bf16)p1[0]; a[5]  = (bf16)p1[1]; a[6]  = (bf16)p1[2]; a[7]  = (bf16)p1[3];
    a[8]  = (bf16)p2[0]; a[9]  = (bf16)p2[1]; a[10] = (bf16)p2[2]; a[11] = (bf16)p2[3];
    a[12] = (bf16)p3[0]; a[13] = (bf16)p3[1]; a[14] = (bf16)p3[2]; a[15] = (bf16)p3[3];

    // ---- B fragments: pre-swizzled, 32 contiguous bytes per lane per tile
    const bf16* bp = zf + ((size_t)((r * KB2 + (k0 >> 5)) * NTILES)) * 512u + lane * 16u;
    v16bf b0 = *(const v16bf*)(bp + 0 * 512);
    v16bf b1 = *(const v16bf*)(bp + 1 * 512);
    v16bf b2 = *(const v16bf*)(bp + 2 * 512);
    v16bf b3 = *(const v16bf*)(bp + 3 * 512);

    acc0 = __builtin_amdgcn_wmma_f32_16x16x32_bf16(false, a, false, b0, (short)0, acc0, false, false);
    acc1 = __builtin_amdgcn_wmma_f32_16x16x32_bf16(false, a, false, b1, (short)0, acc1, false, false);
    acc2 = __builtin_amdgcn_wmma_f32_16x16x32_bf16(false, a, false, b2, (short)0, acc2, false, false);
    acc3 = __builtin_amdgcn_wmma_f32_16x16x32_bf16(false, a, false, b3, (short)0, acc3, false, false);
  }

  // ---- write partial slice (r,kc). C layout: lane 0-15 rows m0+0..7,
  // lane 16-31 rows m0+8..15, col = lane&15 (+16*tile).
  float* pp = partial + ((size_t)(r * KC + kc) * NN) * OUTF;
  const unsigned orow = m0 + khalf * 8u;
  const unsigned ocol = lane & 15u;
#pragma unroll
  for (int i = 0; i < 8; ++i) {
    size_t base = (size_t)(orow + i) * OUTF + ocol;
    pp[base + 0]  = acc0[i];
    pp[base + 16] = acc1[i];
    pp[base + 32] = acc2[i];
    pp[base + 48] = acc3[i];
  }
}

// ---------------------------------------------------------------------------
// Kernel 3: deterministic reduction of the 32 partial slices.
// ---------------------------------------------------------------------------
__global__ void reduce_kernel(const float* __restrict__ partial,
                              float* __restrict__ out) {
  unsigned i = blockIdx.x * blockDim.x + threadIdx.x;  // 0 .. NN*OUTF-1
  float s = 0.0f;
#pragma unroll
  for (int p = 0; p < NSLICE; ++p)
    s += partial[(size_t)p * (NN * OUTF) + i];
  out[i] = s;
}

extern "C" void kernel_launch(void* const* d_in, const int* in_sizes, int n_in,
                              void* d_out, int out_size, void* d_ws, size_t ws_size,
                              hipStream_t stream) {
  const float* adj = (const float*)d_in[0];  // [R, N, N]
  const float* x   = (const float*)d_in[1];  // [N, IN_F]
  const float* w   = (const float*)d_in[2];  // [R, OUT_F, IN_F]
  float* out = (float*)d_out;                // [N, OUT_F]

  bf16*  zf      = (bf16*)d_ws;                                   // 4 MB
  float* partial = (float*)((char*)d_ws +
                            (size_t)RNUM * NN * OUTF * sizeof(bf16));  // 32 MB

  // 1) Z fragments: 8*4096*64 threads
  zfrag_kernel<<<(RNUM * NN * OUTF) / 256, 256, 0, stream>>>(x, w, zf);

  // 2) main GEMM: 32 x 8 x 4 blocks of 8 waves
  dim3 grid(MT, RNUM, KC);
  rgcn_wmma_kernel<<<grid, 256, 0, stream>>>(adj, zf, partial);

  // 3) reduce 32 slices into d_out (writes every output element)
  reduce_kernel<<<(NN * OUTF) / 256, 256, 0, stream>>>(partial, out);
}
